// SelfAttn_16484084483532
// MI455X (gfx1250) — matrix operand order repaired and checked by reading, not compiled, not run on hardware
//
#include <hip/hip_runtime.h>
#include <hip/hip_bf16.h>

typedef __attribute__((ext_vector_type(16))) _Float16 v16h;
typedef __attribute__((ext_vector_type(8)))  _Float16 v8h;
typedef __attribute__((ext_vector_type(8)))  float    v8f;

#define C_CH 256
#define NTOK 1024

// ---------- WMMA helpers ----------

__device__ __forceinline__ v8f wmma_f16(v16h a, v16h b, v8f c) {
  // D = A(16x32 f16) * B(32x16 f16) + C(16x16 f32)
  return __builtin_amdgcn_wmma_f32_16x16x32_f16(false, a, false, b, (short)0, c,
                                                false, false);
}

// A fragment (16x32, M x K) from row-major f16 matrix, row stride ld (elements).
// Lane layout: M = lane%16; K chunks at 8*(lane/16) and 16+8*(lane/16).
__device__ __forceinline__ v16h frag_a_f16(const _Float16* base, int ld, int lane) {
  int m = lane & 15, s = (lane >> 4) * 8;
  const _Float16* r = base + (size_t)m * ld + s;
  v8h a0 = *(const v8h*)(r);
  v8h a1 = *(const v8h*)(r + 16);
  v16h f;
#pragma unroll
  for (int i = 0; i < 8; ++i) { f[i] = a0[i]; f[i + 8] = a1[i]; }
  return f;
}

// A fragment from row-major f32 matrix (LDS scores), converting to f16.
__device__ __forceinline__ v16h frag_a_f32(const float* base, int ld, int lane) {
  int m = lane & 15, s = (lane >> 4) * 8;
  const float* r = base + (size_t)m * ld + s;
  v16h f;
#pragma unroll
  for (int i = 0; i < 8; ++i) {
    f[i]     = (_Float16)r[i];
    f[i + 8] = (_Float16)r[i + 16];
  }
  return f;
}

// B fragment (32x16, K x N) from storage laid out [N][K] row-major, row stride ld.
// Lane layout: N = lane%16; K chunks at 8*(lane/16) and 16+8*(lane/16).
__device__ __forceinline__ v16h frag_b_nk(const _Float16* base, int ld, int lane) {
  const _Float16* r = base + (size_t)(lane & 15) * ld + (lane >> 4) * 8;
  v8h b0 = *(const v8h*)(r);
  v8h b1 = *(const v8h*)(r + 16);
  v16h f;
#pragma unroll
  for (int i = 0; i < 8; ++i) { f[i] = b0[i]; f[i + 8] = b1[i]; }
  return f;
}

__device__ __forceinline__ void zero8(v8f& a) {
#pragma unroll
  for (int i = 0; i < 8; ++i) a[i] = 0.0f;
}

// ---------- Kernel 1: GroupNorm -> Ht [B][N][C] f16 ----------
__global__ void gn_kernel(const float* __restrict__ x, const float* __restrict__ gamma,
                          const float* __restrict__ beta, _Float16* __restrict__ Ht) {
  int bg = blockIdx.x;              // 32 batches * 32 groups
  int b = bg >> 5, g = bg & 31;
  const float* xg = x + (size_t)b * C_CH * NTOK + (size_t)g * 8 * NTOK;
  int t = threadIdx.x;
  int cl = t >> 5;                  // channel within group (0..7)
  int nc = (t & 31) * 32;           // token chunk start
  const float* p = xg + (size_t)cl * NTOK + nc;
  float s = 0.f, s2 = 0.f;
#pragma unroll
  for (int i = 0; i < 32; ++i) { float v = p[i]; s += v; s2 += v * v; }
  __shared__ float red0[256];
  __shared__ float red1[256];
  red0[t] = s; red1[t] = s2;
  __syncthreads();
  for (int off = 128; off > 0; off >>= 1) {
    if (t < off) { red0[t] += red0[t + off]; red1[t] += red1[t + off]; }
    __syncthreads();
  }
  float mean = red0[0] * (1.f / 8192.f);
  float var  = red1[0] * (1.f / 8192.f) - mean * mean;
  float rstd = rsqrtf(var + 1e-5f);
  int c = g * 8 + cl;
  float ga = gamma[c], be = beta[c];
  _Float16* ho = Ht + (size_t)b * NTOK * C_CH + c;
#pragma unroll
  for (int i = 0; i < 32; ++i) {
    float v = (p[i] - mean) * rstd * ga + be;
    ho[(size_t)(nc + i) * C_CH] = (_Float16)v;
  }
}

// ---------- Kernel 2: weight f32 -> f16 (wq|wk|wv|wh concatenated) ----------
__global__ void wcvt_kernel(const float* __restrict__ wq, const float* __restrict__ wk,
                            const float* __restrict__ wv, const float* __restrict__ wh,
                            _Float16* __restrict__ dst) {
  int i = blockIdx.x * 256 + threadIdx.x;   // 0 .. 4*65536-1
  int m = i >> 16, j = i & 65535;
  const float* s;
  if (m == 0) s = wq; else if (m == 1) s = wk; else if (m == 2) s = wv; else s = wh;
  dst[i] = (_Float16)s[j];
}

// ---------- Kernel 3: QKV projection ----------
// Wf: fused [768][256] f16 (Q rows 0..255, K 256..511, V 512..767).
// Q,K -> [B][N][C] f16 ; V -> [B][C][N] f16.
__global__ void qkv_kernel(const _Float16* __restrict__ Wf, const _Float16* __restrict__ Ht,
                           const float* __restrict__ bq, const float* __restrict__ bk,
                           const float* __restrict__ bv,
                           _Float16* __restrict__ Qt, _Float16* __restrict__ Kt,
                           _Float16* __restrict__ Vc) {
  int b  = blockIdx.x >> 6;            // 32 batches * 64 n-tiles
  int n0 = (blockIdx.x & 63) * 16;
  int wave = threadIdx.x >> 5, lane = threadIdx.x & 31;
  const _Float16* Hb = Ht + (size_t)b * NTOK * C_CH + (size_t)n0 * C_CH;
  const _Float16* Wb = Wf + (size_t)(wave * 6) * 16 * C_CH;
  v8f acc[6];
#pragma unroll
  for (int i = 0; i < 6; ++i) zero8(acc[i]);

  // Preload ALL fragments for a k-step into distinct registers, then issue the
  // 6-WMMA burst: one load-clause + one wait per 6 WMMAs instead of per WMMA.
  v16h bf = frag_b_nk(Hb, C_CH, lane);
#pragma unroll
  for (int kk = 0; kk < 8; ++kk) {               // K = 256 in steps of 32
    v16h af[6];
#pragma unroll
    for (int tI = 0; tI < 6; ++tI)
      af[tI] = frag_a_f16(Wb + (size_t)tI * 16 * C_CH + kk * 32, C_CH, lane);
    v16h bf_cur = bf;
    if (kk < 7) bf = frag_b_nk(Hb + (kk + 1) * 32, C_CH, lane);
#pragma unroll
    for (int tI = 0; tI < 6; ++tI)
      acc[tI] = wmma_f16(af[tI], bf_cur, acc[tI]);
  }

  int n = n0 + (lane & 15);
#pragma unroll
  for (int tI = 0; tI < 6; ++tI) {
    int T = wave * 6 + tI;
    int mat = T >> 4;                            // 0=Q 1=K 2=V
    int c0 = (T & 15) * 16 + 8 * (lane >> 4);
    const float* bias;
    if (mat == 0) bias = bq; else if (mat == 1) bias = bk; else bias = bv;
    if (mat < 2) {
      _Float16* dst = ((mat == 0) ? Qt : Kt) +
                      (size_t)b * NTOK * C_CH + (size_t)n * C_CH + c0;
      v8h o;
#pragma unroll
      for (int r = 0; r < 8; ++r) o[r] = (_Float16)(acc[tI][r] + bias[c0 + r]);
      *(v8h*)dst = o;
    } else {
      _Float16* dst = Vc + (size_t)b * C_CH * NTOK;
#pragma unroll
      for (int r = 0; r < 8; ++r)
        dst[(size_t)(c0 + r) * NTOK + n] = (_Float16)(acc[tI][r] + bias[c0 + r]);
    }
  }
}

// ---------- Kernel 4: attention (scores + softmax + P*V), flash-row-block ----------
// One WG = one batch, one 16-row query block. LDS holds S[16][1024] f32 + reduce scratch.
__global__ void attn_kernel(const _Float16* __restrict__ Qt, const _Float16* __restrict__ Kt,
                            const _Float16* __restrict__ Vc, _Float16* __restrict__ Ot) {
  extern __shared__ float smem[];
  float* S   = smem;                  // 16*1024 f32
  float* red = smem + 16 * 1024;      // 2 * 16 * 16 f32
  int b  = blockIdx.x >> 6;
  int n0 = (blockIdx.x & 63) * 16;
  int wave = threadIdx.x >> 5, lane = threadIdx.x & 31;
  const _Float16* Qb = Qt + (size_t)b * NTOK * C_CH + (size_t)n0 * C_CH;
  const _Float16* Kb = Kt + (size_t)b * NTOK * C_CH;
  const _Float16* Vb = Vc + (size_t)b * C_CH * NTOK;

  // Preload the 16x256 Q tile as 8 A-fragments (kept in VGPRs).
  v16h qf[8];
#pragma unroll
  for (int kk = 0; kk < 8; ++kk) qf[kk] = frag_a_f16(Qb + kk * 32, C_CH, lane);

  // Phase A: S[n][m] = (Q^T K)/sqrt(C).
  // Each wave owns 8 of the 64 m-tiles, processed 2 at a time; k-loop is
  // unrolled by 2 so each load-clause feeds a 4-WMMA burst.
  for (int j = 0; j < 4; ++j) {
    int mt0 = (wave * 2 + j * 16) * 16;          // first m-tile row of the pair
    if (j < 3) {
      __builtin_prefetch(Kb + (size_t)(mt0 + 256) * C_CH + lane * 8, 0, 1);
      __builtin_prefetch(Kb + (size_t)(mt0 + 272) * C_CH + lane * 8, 0, 1);
    }
    v8f acc0, acc1; zero8(acc0); zero8(acc1);
#pragma unroll
    for (int kk = 0; kk < 8; kk += 2) {
      v16h k00 = frag_b_nk(Kb + (size_t)mt0 * C_CH + kk * 32, C_CH, lane);
      v16h k10 = frag_b_nk(Kb + (size_t)(mt0 + 16) * C_CH + kk * 32, C_CH, lane);
      v16h k01 = frag_b_nk(Kb + (size_t)mt0 * C_CH + (kk + 1) * 32, C_CH, lane);
      v16h k11 = frag_b_nk(Kb + (size_t)(mt0 + 16) * C_CH + (kk + 1) * 32, C_CH, lane);
      acc0 = wmma_f16(qf[kk], k00, acc0);
      acc1 = wmma_f16(qf[kk], k10, acc1);
      acc0 = wmma_f16(qf[kk + 1], k01, acc0);
      acc1 = wmma_f16(qf[kk + 1], k11, acc1);
    }
    int m = mt0 + (lane & 15);
    int rbase = 8 * (lane >> 4);
#pragma unroll
    for (int r = 0; r < 8; ++r) {
      S[(size_t)(rbase + r) * 1024 + m]      = acc0[r] * 0.0625f;
      S[(size_t)(rbase + r) * 1024 + m + 16] = acc1[r] * 0.0625f;
    }
  }
  __syncthreads();

  // Block softmax over each of the 16 rows (16 threads per row, 64 elems each).
  {
    int t = threadIdx.x;
    int row = t >> 4, seg = t & 15;
    float* Sr = S + (size_t)row * 1024 + seg * 64;
    float mx = -1e30f;
#pragma unroll
    for (int i = 0; i < 64; ++i) mx = fmaxf(mx, Sr[i]);
    red[row * 16 + seg] = mx;
    __syncthreads();
    float rmx = -1e30f;
#pragma unroll
    for (int i = 0; i < 16; ++i) rmx = fmaxf(rmx, red[row * 16 + i]);
    float sum = 0.f;
#pragma unroll
    for (int i = 0; i < 64; ++i) { float e = __expf(Sr[i] - rmx); Sr[i] = e; sum += e; }
    red[256 + row * 16 + seg] = sum;
    __syncthreads();
    float rs = 0.f;
#pragma unroll
    for (int i = 0; i < 16; ++i) rs += red[256 + row * 16 + i];
    float inv = 1.f / rs;
#pragma unroll
    for (int i = 0; i < 64; ++i) Sr[i] *= inv;
  }
  __syncthreads();

  // Phase B: O^T[n][c] = P * V^T ; c-dimension split across waves (32 cols each).
  // m-loop unrolled by 2: 4 global V-frags + 2 LDS P-frags per 4-WMMA burst.
  v8f oacc[2];
  zero8(oacc[0]); zero8(oacc[1]);
  int c0 = wave * 32;
  for (int mk = 0; mk < 32; mk += 2) {           // K = 1024 (m) in steps of 64
    v16h vf00 = frag_b_nk(Vb + (size_t)c0 * NTOK + mk * 32, NTOK, lane);
    v16h vf10 = frag_b_nk(Vb + (size_t)(c0 + 16) * NTOK + mk * 32, NTOK, lane);
    v16h vf01 = frag_b_nk(Vb + (size_t)c0 * NTOK + (mk + 1) * 32, NTOK, lane);
    v16h vf11 = frag_b_nk(Vb + (size_t)(c0 + 16) * NTOK + (mk + 1) * 32, NTOK, lane);
    v16h pf0 = frag_a_f32(S + mk * 32, 1024, lane);
    v16h pf1 = frag_a_f32(S + (mk + 1) * 32, 1024, lane);
    oacc[0] = wmma_f16(pf0, vf00, oacc[0]);
    oacc[1] = wmma_f16(pf0, vf10, oacc[1]);
    oacc[0] = wmma_f16(pf1, vf01, oacc[0]);
    oacc[1] = wmma_f16(pf1, vf11, oacc[1]);
  }
  _Float16* Ob = Ot + (size_t)b * NTOK * C_CH;
#pragma unroll
  for (int ci = 0; ci < 2; ++ci) {
    int c  = c0 + ci * 16 + (lane & 15);
    int nb = n0 + 8 * (lane >> 4);
#pragma unroll
    for (int r = 0; r < 8; ++r)
      Ob[(size_t)(nb + r) * C_CH + c] = (_Float16)oacc[ci][r];
  }
}

// ---------- Kernel 5: output projection + bias + residual ----------
__global__ void proj_kernel(const _Float16* __restrict__ Whf, const _Float16* __restrict__ Ot,
                            const float* __restrict__ bh, const float* __restrict__ x,
                            float* __restrict__ out) {
  int b  = blockIdx.x >> 6;
  int n0 = (blockIdx.x & 63) * 16;
  int wave = threadIdx.x >> 5, lane = threadIdx.x & 31;
  const _Float16* Ob = Ot + (size_t)b * NTOK * C_CH + (size_t)n0 * C_CH;
  const _Float16* Wb = Whf + (size_t)(wave * 2) * 16 * C_CH;
  v8f acc[2];
  zero8(acc[0]); zero8(acc[1]);
#pragma unroll
  for (int kk = 0; kk < 8; kk += 2) {            // k unrolled by 2 -> 4-WMMA bursts
    v16h a00 = frag_a_f16(Wb + kk * 32, C_CH, lane);
    v16h a10 = frag_a_f16(Wb + (size_t)16 * C_CH + kk * 32, C_CH, lane);
    v16h a01 = frag_a_f16(Wb + (kk + 1) * 32, C_CH, lane);
    v16h a11 = frag_a_f16(Wb + (size_t)16 * C_CH + (kk + 1) * 32, C_CH, lane);
    v16h b0 = frag_b_nk(Ob + kk * 32, C_CH, lane);
    v16h b1 = frag_b_nk(Ob + (kk + 1) * 32, C_CH, lane);
    acc[0] = wmma_f16(a00, b0, acc[0]);
    acc[1] = wmma_f16(a10, b0, acc[1]);
    acc[0] = wmma_f16(a01, b1, acc[0]);
    acc[1] = wmma_f16(a11, b1, acc[1]);
  }
  int n = n0 + (lane & 15);
  const float* xb = x + (size_t)b * C_CH * NTOK;
  float* ob = out + (size_t)b * C_CH * NTOK;
#pragma unroll
  for (int tI = 0; tI < 2; ++tI) {
    int T = wave * 2 + tI;
    int c0 = T * 16 + 8 * (lane >> 4);
#pragma unroll
    for (int r = 0; r < 8; ++r) {
      int c = c0 + r;
      size_t idx = (size_t)c * NTOK + n;
      ob[idx] = xb[idx] + acc[tI][r] + bh[c];
    }
  }
}

// ---------- Launch ----------
extern "C" void kernel_launch(void* const* d_in, const int* in_sizes, int n_in,
                              void* d_out, int out_size, void* d_ws, size_t ws_size,
                              hipStream_t stream) {
  const float* x     = (const float*)d_in[0];
  const float* gamma = (const float*)d_in[1];
  const float* beta  = (const float*)d_in[2];
  const float* wq    = (const float*)d_in[3];
  const float* bq    = (const float*)d_in[4];
  const float* wk    = (const float*)d_in[5];
  const float* bk    = (const float*)d_in[6];
  const float* wv    = (const float*)d_in[7];
  const float* bv    = (const float*)d_in[8];
  const float* wh    = (const float*)d_in[9];
  const float* bh    = (const float*)d_in[10];
  float* out = (float*)d_out;

  char* ws = (char*)d_ws;
  _Float16* Wf = (_Float16*)ws;                 // 4 * 256*256 f16 = 512 KB
  _Float16* Ht = (_Float16*)(ws + 524288);      // each buffer: 32*1024*256 f16 = 16 MB
  _Float16* Qt = Ht + (size_t)8388608;
  _Float16* Kt = Qt + (size_t)8388608;
  _Float16* Vc = Kt + (size_t)8388608;
  _Float16* Ot = Vc + (size_t)8388608;

  wcvt_kernel<<<1024, 256, 0, stream>>>(wq, wk, wv, wh, Wf);
  gn_kernel<<<1024, 256, 0, stream>>>(x, gamma, beta, Ht);
  qkv_kernel<<<2048, 256, 0, stream>>>(Wf, Ht, bq, bk, bv, Qt, Kt, Vc);
  attn_kernel<<<2048, 256, 16 * 1024 * 4 + 512 * 4, stream>>>(Qt, Kt, Vc, Ot);
  proj_kernel<<<2048, 256, 0, stream>>>(Wf + (size_t)3 * 65536, Ot, bh, x, out);
}
